// XEGNNK_80272938762991
// MI455X (gfx1250) — matrix operand order repaired and checked by reading, not compiled, not run on hardware
//
#include <hip/hip_runtime.h>
#include <hip/hip_bf16.h>
#include <math.h>

#define NN 20000
#define EE 320000
#define DD 64
#define KK 128
#define GG 64
#define CLAMPV 10.0f

typedef __attribute__((ext_vector_type(16))) __bf16 v16bf;
typedef __attribute__((ext_vector_type(8)))  __bf16 v8bf;
typedef __attribute__((ext_vector_type(8)))  float  v8f;

// ---------- workspace layout (256B aligned regions) ----------
static constexpr size_t OFF_GSUM = 0;                       // 98304 B
static constexpr size_t OFF_GCNT = 98304;                   // 256 B
static constexpr size_t OFF_NSUM = 98560;                   // 32768 B
static constexpr size_t OFF_ACC_END = 131328;               // zeroed each call
static constexpr size_t OFF_XN   = 131328;                  // 30,720,000 B
static constexpr size_t OFF_HN   = 30851328;                // 5,120,000 B
static constexpr size_t OFF_PW1  = 35971328;                // 49152 B
static constexpr size_t OFF_PW2  = 36020480;                // 8192 B
static constexpr size_t OFF_PX1  = 36028672;                // 8192 B
static constexpr size_t OFF_PX2  = 36036864;                // 16384 B

static __device__ __forceinline__ __bf16 to_bf16(float f) {
  unsigned int u = __builtin_bit_cast(unsigned int, f);
  unsigned int r = (u + 0x7FFFu + ((u >> 16) & 1u)) >> 16;
  unsigned short s = (unsigned short)r;
  return __builtin_bit_cast(__bf16, s);
}

// 16x32 bf16 A-fragment from an LDS row (two 16B ds reads per lane).
// Lane L: M=L%16, h=L/16 ; elem j<8 -> K=base+j ; j>=8 -> K=base+16+(j-8).
static __device__ __forceinline__ v16bf ldsA(const __bf16* row, int base) {
  union { v16bf v; v8bf h[2]; } u;
  u.h[0] = *(const v8bf*)(row + base);
  u.h[1] = *(const v8bf*)(row + base + 16);
  return u.v;
}

static __device__ __forceinline__ v8f wmma_bf16(v16bf a, v16bf b, v8f c) {
  return __builtin_amdgcn_wmma_f32_16x16x32_bf16(false, a, false, b, (short)0, c,
                                                 false, false);
}

// ---------------- per-graph sum of X + node counts ----------------
__global__ __launch_bounds__(256) void k_gsum(const float* __restrict__ X,
                                              const int* __restrict__ batch,
                                              float* __restrict__ gsum,
                                              float* __restrict__ gcnt) {
  int id = blockIdx.x * 256 + threadIdx.x;
  if (id >= NN * 384) return;
  int n = id / 384, j = id - n * 384;
  int g = batch[n];
  atomicAdd(&gsum[g * 384 + j], X[id]);
  if (j == 0) atomicAdd(&gcnt[g], 1.0f);
}

// ---------------- center X, accumulate per-graph norms ----------------
__global__ __launch_bounds__(256) void k_center(const float* __restrict__ X,
                                                const int* __restrict__ batch,
                                                const float* __restrict__ gsum,
                                                const float* __restrict__ gcnt,
                                                float* __restrict__ nsum,
                                                float* __restrict__ Xc) {
  int id = blockIdx.x * 256 + threadIdx.x;
  if (id >= NN * KK) return;
  int n = id / KK, k = id - n * KK;
  int g = batch[n];
  float inv = 1.f / fmaxf(gcnt[g], 1.f);
  float s = 0.f;
#pragma unroll
  for (int d = 0; d < 3; ++d) {
    int idx = n * 384 + d * KK + k;
    float v = X[idx] - gsum[g * 384 + d * KK + k] * inv;
    Xc[idx] = v;
    s += v * v;
  }
  atomicAdd(&nsum[g * KK + k], sqrtf(s));
}

// ---------------- E3 normalize; write Xn and init out = Xn ----------------
__global__ __launch_bounds__(256) void k_normalize(const int* __restrict__ batch,
                                                   const float* __restrict__ gcnt,
                                                   const float* __restrict__ nsum,
                                                   const float* __restrict__ xnw,
                                                   float* __restrict__ Xn,
                                                   float* __restrict__ out) {
  int id = blockIdx.x * 256 + threadIdx.x;
  if (id >= NN * KK) return;
  int n = id / KK, k = id - n * KK;
  int g = batch[n];
  float inv = 1.f / fmaxf(gcnt[g], 1.f);
  float mn = nsum[g * KK + k] * inv;
  float sc = xnw[k] / (mn + 1e-5f);
#pragma unroll
  for (int d = 0; d < 3; ++d) {
    int idx = n * 384 + d * KK + k;
    float v = Xn[idx] * sc;
    Xn[idx] = v;
    out[idx] = v;
  }
}

// ---------------- LayerNorm H -> Hn (one wave32 per row) ----------------
__global__ __launch_bounds__(128) void k_ln(const float* __restrict__ H,
                                            const float* __restrict__ g,
                                            const float* __restrict__ b,
                                            float* __restrict__ Hn) {
  int row = blockIdx.x * 4 + (threadIdx.x >> 5);
  int lane = threadIdx.x & 31;
  if (row >= NN) return;
  float a0 = H[row * DD + lane];
  float a1 = H[row * DD + lane + 32];
  float s = a0 + a1, q = a0 * a0 + a1 * a1;
#pragma unroll
  for (int i = 16; i > 0; i >>= 1) {
    s += __shfl_xor(s, i, 32);
    q += __shfl_xor(q, i, 32);
  }
  float mean = s * (1.f / 64.f);
  float var = q * (1.f / 64.f) - mean * mean;
  float rstd = rsqrtf(var + 1e-5f);
  Hn[row * DD + lane]      = (a0 - mean) * rstd * g[lane]      + b[lane];
  Hn[row * DD + lane + 32] = (a1 - mean) * rstd * g[lane + 32] + b[lane + 32];
}

// ---------------- pack row-major W[K,N] into WMMA bf16 B-fragments ----------
// Consumer reads v16bf at ((chunk*NT + nt)*32 + lane); lane L: Ncol = nt*16+L%16,
// element j: Krow = chunk*32 + (L/16)*8 + (j&7) + (j>>3)*16.
__global__ __launch_bounds__(256) void k_pack(const float* __restrict__ W,
                                              int Kdim, int Ndim,
                                              __bf16* __restrict__ outp) {
  int id = blockIdx.x * 256 + threadIdx.x;
  if (id >= Kdim * Ndim) return;
  int j = id & 15;
  int lane = (id >> 4) & 31;
  int rest = id >> 9;
  int NT = Ndim >> 4;
  int nt = rest % NT;
  int chunk = rest / NT;
  int kk = chunk * 32 + (lane >> 4) * 8 + (j & 7) + (j >> 3) * 16;
  int nn = nt * 16 + (lane & 15);
  outp[id] = to_bf16(W[kk * Ndim + nn]);
}

// ---------------- fused edge pipeline: 32 edges / block, 8 waves ------------
// GEMM1..3: one 16x16 output tile per wave (mt=w>>2, nt=w&3).
// GEMM4: wave w owns N-tile w for BOTH M-tiles (B-fragment reuse).
__global__ __launch_bounds__(256) void k_edges(
    const float* __restrict__ Xn, const float* __restrict__ Hn,
    const float* __restrict__ edge_attr, const float* __restrict__ te,
    const __bf16* __restrict__ pWm1, const float* __restrict__ bm1,
    const __bf16* __restrict__ pWm2, const float* __restrict__ bm2,
    const __bf16* __restrict__ pWx1, const float* __restrict__ bx1,
    const __bf16* __restrict__ pWx2, const float* __restrict__ bx2,
    const int* __restrict__ eidx, float* __restrict__ out) {
  __shared__ __attribute__((aligned(16))) __bf16 sA[32][392];   // msg tile (pad=8)
  __shared__ __attribute__((aligned(16))) __bf16 sHa[32][72];   // pad=8
  __shared__ __attribute__((aligned(16))) __bf16 sHb[32][72];
  __shared__ float sW[32][KK];
  __shared__ int sSrc[32], sTgt[32];

  const int tid = threadIdx.x;
  const int e0 = blockIdx.x * 32;
  {
    const int r = tid >> 3, p = tid & 7;
    if (p == 0) { sSrc[r] = eidx[e0 + r]; sTgt[r] = eidx[EE + e0 + r]; }
  }
  __syncthreads();
  { // gather: 8 threads per edge build the 384-col bf16 message row
    const int r = tid >> 3, p = tid & 7;
    const int e = e0 + r;
    const int sn = sSrc[r], tn = sTgt[r];
    const int f0 = p * 8;
    v8bf t0, t1, t2, t3;
#pragma unroll
    for (int i = 0; i < 8; ++i) {
      t0[i] = to_bf16(Hn[tn * DD + f0 + i]);          // Hn[tgt]
      t1[i] = to_bf16(Hn[sn * DD + f0 + i]);          // Hn[src]
      t2[i] = to_bf16(edge_attr[e * DD + f0 + i]);
      t3[i] = to_bf16(te[e * DD + f0 + i]);
    }
    *(v8bf*)&sA[r][f0]       = t0;
    *(v8bf*)&sA[r][64 + f0]  = t1;
    *(v8bf*)&sA[r][128 + f0] = t2;
    *(v8bf*)&sA[r][320 + f0] = t3;
    const int k0 = p * 16;
#pragma unroll
    for (int c = 0; c < 2; ++c) {
      v8bf td;
#pragma unroll
      for (int i = 0; i < 8; ++i) {
        int k = k0 + c * 8 + i;
        float rx = Xn[sn * 384 + k]            - Xn[tn * 384 + k];
        float ry = Xn[sn * 384 + KK + k]       - Xn[tn * 384 + KK + k];
        float rz = Xn[sn * 384 + 2 * KK + k]   - Xn[tn * 384 + 2 * KK + k];
        td[i] = to_bf16(rx * rx + ry * ry + rz * rz);
      }
      *(v8bf*)&sA[r][192 + k0 + c * 8] = td;
    }
  }
  __syncthreads();

  const int w = tid >> 5, lane = tid & 31;
  const int lm = lane & 15, lh = lane >> 4;
  const int mt = w >> 2, nt = w & 3;
  const int arow = lm + 16 * mt;   // A row for GEMM1..3
  const int mrow0 = lh * 8 + 16 * mt;

  // GEMM1: h1 = silu([32,384] @ Wm1 + bm1)
  {
    v8f acc;
#pragma unroll
    for (int z = 0; z < 8; ++z) acc[z] = 0.f;
#pragma unroll
    for (int chunk = 0; chunk < 12; ++chunk) {
      v16bf a = ldsA(&sA[arow][0], chunk * 32 + lh * 8);
      v16bf b = ((const v16bf*)pWm1)[(chunk * 4 + nt) * 32 + lane];
      acc = wmma_bf16(a, b, acc);
    }
    const float bias = bm1[nt * 16 + lm];
#pragma unroll
    for (int rr = 0; rr < 8; ++rr) {
      float v = acc[rr] + bias;
      float s = v / (1.f + __expf(-v));                      // SiLU
      sHa[rr + mrow0][nt * 16 + lm] = to_bf16(s);
    }
  }
  __syncthreads();

  // GEMM2: m_ij = h1 @ Wm2 + bm2
  {
    v8f acc;
#pragma unroll
    for (int z = 0; z < 8; ++z) acc[z] = 0.f;
#pragma unroll
    for (int chunk = 0; chunk < 2; ++chunk) {
      v16bf a = ldsA(&sHa[arow][0], chunk * 32 + lh * 8);
      v16bf b = ((const v16bf*)pWm2)[(chunk * 4 + nt) * 32 + lane];
      acc = wmma_bf16(a, b, acc);
    }
    const float bias = bm2[nt * 16 + lm];
#pragma unroll
    for (int rr = 0; rr < 8; ++rr)
      sHb[rr + mrow0][nt * 16 + lm] = to_bf16(acc[rr] + bias);
  }
  __syncthreads();

  // GEMM3: h2 = silu(m_ij @ Wx1 + bx1)
  {
    v8f acc;
#pragma unroll
    for (int z = 0; z < 8; ++z) acc[z] = 0.f;
#pragma unroll
    for (int chunk = 0; chunk < 2; ++chunk) {
      v16bf a = ldsA(&sHb[arow][0], chunk * 32 + lh * 8);
      v16bf b = ((const v16bf*)pWx1)[(chunk * 4 + nt) * 32 + lane];
      acc = wmma_bf16(a, b, acc);
    }
    const float bias = bx1[nt * 16 + lm];
#pragma unroll
    for (int rr = 0; rr < 8; ++rr) {
      float v = acc[rr] + bias;
      float s = v / (1.f + __expf(-v));
      sHa[rr + mrow0][nt * 16 + lm] = to_bf16(s);
    }
  }
  __syncthreads();

  // GEMM4: w = clamp(h2 @ Wx2 + bx2); wave w = N-tile, both M-tiles share B.
  {
    v8f acc0, acc1;
#pragma unroll
    for (int z = 0; z < 8; ++z) { acc0[z] = 0.f; acc1[z] = 0.f; }
#pragma unroll
    for (int chunk = 0; chunk < 2; ++chunk) {
      v16bf b = ((const v16bf*)pWx2)[(chunk * 8 + w) * 32 + lane];
      v16bf a0 = ldsA(&sHa[lm][0],      chunk * 32 + lh * 8);
      v16bf a1 = ldsA(&sHa[lm + 16][0], chunk * 32 + lh * 8);
      acc0 = wmma_bf16(a0, b, acc0);
      acc1 = wmma_bf16(a1, b, acc1);
    }
    const float bias = bx2[w * 16 + lm];
#pragma unroll
    for (int rr = 0; rr < 8; ++rr) {
      float v0 = fminf(CLAMPV, fmaxf(-CLAMPV, acc0[rr] + bias));
      float v1 = fminf(CLAMPV, fmaxf(-CLAMPV, acc1[rr] + bias));
      sW[rr + lh * 8][w * 16 + lm]      = v0;
      sW[rr + lh * 8 + 16][w * 16 + lm] = v1;
    }
  }
  __syncthreads();

  { // scatter: recompute rel / rel_dist (L2-hot) and atomically add to out[tgt]
    const int r = tid >> 3, p = tid & 7;
    const int sn = sSrc[r], tn = sTgt[r];
#pragma unroll
    for (int i = 0; i < 16; ++i) {
      int k = p * 16 + i;
      float rx = Xn[sn * 384 + k]          - Xn[tn * 384 + k];
      float ry = Xn[sn * 384 + KK + k]     - Xn[tn * 384 + KK + k];
      float rz = Xn[sn * 384 + 2 * KK + k] - Xn[tn * 384 + 2 * KK + k];
      float rd = rx * rx + ry * ry + rz * rz;
      float f = sW[r][k] / (1.f + sqrtf(rd + 1e-8f));
      atomicAdd(&out[tn * 384 + k],          rx * f);
      atomicAdd(&out[tn * 384 + KK + k],     ry * f);
      atomicAdd(&out[tn * 384 + 2 * KK + k], rz * f);
    }
  }
}

extern "C" void kernel_launch(void* const* d_in, const int* in_sizes, int n_in,
                              void* d_out, int out_size, void* d_ws, size_t ws_size,
                              hipStream_t stream) {
  const float* X   = (const float*)d_in[0];
  const float* H   = (const float*)d_in[1];
  const float* EA  = (const float*)d_in[2];
  const float* TE  = (const float*)d_in[3];
  const float* Wm1 = (const float*)d_in[4];
  const float* bm1 = (const float*)d_in[5];
  const float* Wm2 = (const float*)d_in[6];
  const float* bm2 = (const float*)d_in[7];
  const float* Wx1 = (const float*)d_in[8];
  const float* bx1 = (const float*)d_in[9];
  const float* Wx2 = (const float*)d_in[10];
  const float* bx2 = (const float*)d_in[11];
  const float* lng = (const float*)d_in[12];
  const float* lnb = (const float*)d_in[13];
  const float* xnw = (const float*)d_in[14];
  const int* batch = (const int*)d_in[15];
  const int* eidx  = (const int*)d_in[16];
  float* out = (float*)d_out;
  char* ws = (char*)d_ws;

  float* gsum = (float*)(ws + OFF_GSUM);
  float* gcnt = (float*)(ws + OFF_GCNT);
  float* nsum = (float*)(ws + OFF_NSUM);
  float* Xn   = (float*)(ws + OFF_XN);
  float* Hn   = (float*)(ws + OFF_HN);
  __bf16* pW1 = (__bf16*)(ws + OFF_PW1);
  __bf16* pW2 = (__bf16*)(ws + OFF_PW2);
  __bf16* pX1 = (__bf16*)(ws + OFF_PX1);
  __bf16* pX2 = (__bf16*)(ws + OFF_PX2);

  hipMemsetAsync(ws, 0, OFF_ACC_END, stream);

  k_gsum<<<(NN * 384 + 255) / 256, 256, 0, stream>>>(X, batch, gsum, gcnt);
  k_center<<<(NN * KK + 255) / 256, 256, 0, stream>>>(X, batch, gsum, gcnt, nsum, Xn);
  k_normalize<<<(NN * KK + 255) / 256, 256, 0, stream>>>(batch, gcnt, nsum, xnw, Xn, out);
  k_ln<<<(NN + 3) / 4, 128, 0, stream>>>(H, lng, lnb, Hn);
  k_pack<<<(384 * 64 + 255) / 256, 256, 0, stream>>>(Wm1, 384, 64, pW1);
  k_pack<<<(64 * 64 + 255) / 256, 256, 0, stream>>>(Wm2, 64, 64, pW2);
  k_pack<<<(64 * 64 + 255) / 256, 256, 0, stream>>>(Wx1, 64, 64, pX1);
  k_pack<<<(64 * 128 + 255) / 256, 256, 0, stream>>>(Wx2, 64, 128, pX2);
  k_edges<<<EE / 32, 256, 0, stream>>>(Xn, Hn, EA, TE, pW1, bm1, pW2, bm2,
                                       pX1, bx1, pX2, bx2, eidx, out);
}
